// HybridKernelEstimator_65481071399956
// MI455X (gfx1250) — matrix operand order repaired and checked by reading, not compiled, not run on hardware
//
#include <hip/hip_runtime.h>
#include <hip/hip_bf16.h>

// ---------------------------------------------------------------------------
// HybridKernelEstimator for MI455X (gfx1250)
//   K   = exp(-g * (|x|^2 + |xt|^2 - 2 x.xt))   [4096 x 8192], never in HBM
//   out = tanh(tanh(K@W1 + b1)@W2 + b2)@W3 + b3
// Main GEMM: v_wmma_f32_16x16x32_bf16, fed by async global->LDS copies
// (ASYNCcnt) with double-buffered software pipelining.
// ---------------------------------------------------------------------------

typedef __attribute__((ext_vector_type(16))) __bf16 v16bf;
typedef __attribute__((ext_vector_type(8)))  float  v8f;
typedef __attribute__((ext_vector_type(4)))  int    v4i;

#define GAMMA_F 0.001f
#define NTEST   4096
#define NTRAIN  8192
#define DDIM    512

static __device__ __forceinline__ unsigned short f32_to_bf16_rne(float f) {
    unsigned int u = __float_as_uint(f);
    unsigned int r = u + 0x7FFFu + ((u >> 16) & 1u);   // round-to-nearest-even
    return (unsigned short)(r >> 16);
}

// ---- CDNA5 async global->LDS (tracked by ASYNCcnt) ------------------------
// Builtin prototype (from clang diagnostic): (int4 AS1*, int4 AS3*, Iimm, Iimm)
static __device__ __forceinline__ void async_load_b128(const unsigned short* gsrc,
                                                       unsigned short* ldst) {
#if __has_builtin(__builtin_amdgcn_global_load_async_to_lds_b128)
    __builtin_amdgcn_global_load_async_to_lds_b128(
        (__attribute__((address_space(1))) v4i*)gsrc,
        (__attribute__((address_space(3))) v4i*)ldst,
        0, 0);
#else
    asm volatile("global_load_async_to_lds_b128 %0, %1, off"
                 :: "v"(ldst), "v"(gsrc) : "memory");
#endif
}

static __device__ __forceinline__ void wait_async0() {
#if __has_builtin(__builtin_amdgcn_s_wait_asynccnt)
    __builtin_amdgcn_s_wait_asynccnt(0);
#else
    asm volatile("s_wait_asynccnt 0x0" ::: "memory");
#endif
}

// ---------------------------------------------------------------------------
// Kernel A: f32 -> bf16 conversion + row |x|^2 norms.  One block per row,
// 128 threads, 4 elements each (d = 512).
// ---------------------------------------------------------------------------
__global__ __launch_bounds__(128)
void cvt_norm_kernel(const float* __restrict__ src,
                     unsigned short* __restrict__ dst,
                     float* __restrict__ norms) {
    __shared__ float red[128];
    const int row = blockIdx.x;
    const int t   = threadIdx.x;                 // 0..127
    const float4 v = *(const float4*)(src + (size_t)row * DDIM + t * 4);

    uint2 pk;
    pk.x = (unsigned)f32_to_bf16_rne(v.x) | ((unsigned)f32_to_bf16_rne(v.y) << 16);
    pk.y = (unsigned)f32_to_bf16_rne(v.z) | ((unsigned)f32_to_bf16_rne(v.w) << 16);
    *(uint2*)(dst + (size_t)row * DDIM + t * 4) = pk;

    red[t] = v.x * v.x + v.y * v.y + v.z * v.z + v.w * v.w;
    __syncthreads();
    for (int off = 64; off > 0; off >>= 1) {
        if (t < off) red[t] += red[t + off];
        __syncthreads();
    }
    if (t == 0) norms[row] = red[0];
}

// ---------------------------------------------------------------------------
// Kernel B: fused RBF-GEMM + K@W1 partial contraction.
// grid = (NTEST/128, 16 splits); block = 256 threads (8 waves of 32).
// Block tile: 128 test rows x 512 train cols (8 sub-tiles of 64 cols).
// Wave (wm 0..1, wn 0..3) owns a 64x16 region = 4 WMMA accumulators.
// A/B tiles double-buffered in LDS, filled by async global->LDS DMA.
// ---------------------------------------------------------------------------
__global__ __launch_bounds__(256)
void rbf_wmma_kernel(const unsigned short* __restrict__ Xb,    // [4096][512] bf16
                     const unsigned short* __restrict__ XTb,   // [8192][512] bf16
                     const float* __restrict__ nX,             // [4096]
                     const float* __restrict__ nXT,            // [8192]
                     const float* __restrict__ W1,             // [8192][8]
                     float* __restrict__ partial)              // [16][4096][8]
{
    __shared__ unsigned short As[2][128][40];  // 128x32 bf16 tile x2, stride 40
    __shared__ unsigned short Bs[2][64][40];   // 64x32  bf16 tile x2
    __shared__ float          Ks[128][69];     // K tile, stride 69 (odd -> no conflicts)
    __shared__ float          W1s[64][8];
    __shared__ float          nXs[128];
    __shared__ float          nXTs[64];

    const int tid   = threadIdx.x;
    const int lane  = tid & 31;
    const int wave  = tid >> 5;
    const int wm    = wave >> 2;        // 0..1  (64-row half)
    const int wn    = wave & 3;         // 0..3  (16-col quarter of 64)
    const int lhalf = lane >> 4;        // 0/1
    const int l16   = lane & 15;

    const int m0    = blockIdx.x * 128;
    const int split = blockIdx.y;       // 0..15
    const int n0s   = split * 512;

    if (tid < 128) nXs[tid] = nX[m0 + tid];

    // Per-thread staging assignments (constant across iterations).
    const int arow = tid >> 1;
    const int akc  = (tid & 1) * 16;
    const int brow = tid >> 2;
    const int bkc  = (tid & 3) * 8;

    const int hm = tid & 127;           // row this thread contracts in h-dot phase
    const int jh = tid >> 7;            // 0/1 -> j in [0,4) or [4,8)
    float hreg[4] = {0.f, 0.f, 0.f, 0.f};

    for (int tile = 0; tile < 8; ++tile) {
        const int nbase = n0s + tile * 64;
        __syncthreads();                              // protect W1s/nXTs/Ks/buffers
        if (tid < 64) nXTs[tid] = nXT[nbase + tid];
        {   // stage W1 slice [64][8] = 512 floats, 2 per thread
            const int idx = tid * 2;
            *(float2*)(&((float*)W1s)[idx]) =
                *(const float2*)(&W1[(size_t)nbase * 8 + idx]);
        }

        v8f acc[4];
        acc[0] = (v8f)(0.f); acc[1] = (v8f)(0.f);
        acc[2] = (v8f)(0.f); acc[3] = (v8f)(0.f);

        // Prologue: kick off async DMA for k-step 0 into buffer 0.
        {
            const unsigned short* ga = Xb + (size_t)(m0 + arow) * DDIM + akc;
            async_load_b128(ga,     &As[0][arow][akc]);
            async_load_b128(ga + 8, &As[0][arow][akc + 8]);
            const unsigned short* gb = XTb + (size_t)(nbase + brow) * DDIM + bkc;
            async_load_b128(gb, &Bs[0][brow][bkc]);
        }

        for (int it = 0; it < DDIM / 32; ++it) {
            const int cur = it & 1;
            wait_async0();                 // my async writes into buf[cur] landed
            __syncthreads();               // everyone's landed; prev buf fully read

            if (it + 1 < DDIM / 32) {      // pipeline: DMA next k-step into buf^1
                const int k1 = (it + 1) * 32;
                const unsigned short* ga = Xb + (size_t)(m0 + arow) * DDIM + k1 + akc;
                async_load_b128(ga,     &As[cur ^ 1][arow][akc]);
                async_load_b128(ga + 8, &As[cur ^ 1][arow][akc + 8]);
                const unsigned short* gb = XTb + (size_t)(nbase + brow) * DDIM + k1 + bkc;
                async_load_b128(gb, &Bs[cur ^ 1][brow][bkc]);
            }

            // B fragment: col n = wn*16 + l16, chunks at K = 8*lhalf and 16 + 8*lhalf
            union { v16bf v; uint4 u[2]; } bf;
            {
                const unsigned short* p = &Bs[cur][wn * 16 + l16][8 * lhalf];
                bf.u[0] = *(const uint4*)p;
                bf.u[1] = *(const uint4*)(p + 16);
            }
            // A fragments: row m, chunks at K = 8*lhalf and 16 + 8*lhalf
            #pragma unroll
            for (int mt = 0; mt < 4; ++mt) {
                union { v16bf v; uint4 u[2]; } af;
                const unsigned short* p = &As[cur][wm * 64 + mt * 16 + l16][8 * lhalf];
                af.u[0] = *(const uint4*)p;
                af.u[1] = *(const uint4*)(p + 16);
                acc[mt] = __builtin_amdgcn_wmma_f32_16x16x32_bf16(
                    false, af.v, false, bf.v, (short)0, acc[mt], false, false);
            }
        }

        __syncthreads();
        // Epilogue: S -> K = exp(-g*max(a2+b2-2S,0)), into LDS K tile.
        {
            const int nl = wn * 16 + l16;
            const float bn = nXTs[nl];
            #pragma unroll
            for (int mt = 0; mt < 4; ++mt) {
                const int mbase = wm * 64 + mt * 16 + 8 * lhalf;
                #pragma unroll
                for (int c = 0; c < 8; ++c) {
                    const int m = mbase + c;
                    float sq = nXs[m] + bn - 2.0f * acc[mt][c];
                    sq = fmaxf(sq, 0.0f);
                    Ks[m][nl] = __expf(-GAMMA_F * sq);
                }
            }
        }
        __syncthreads();
        // h-dot: hreg[j] += sum_n K[m][n] * W1[n][j]   (64-deep, tiny VALU pass)
        {
            const int jb = jh * 4;
            #pragma unroll 4
            for (int n = 0; n < 64; ++n) {
                const float kv = Ks[hm][n];
                hreg[0] = fmaf(kv, W1s[n][jb + 0], hreg[0]);
                hreg[1] = fmaf(kv, W1s[n][jb + 1], hreg[1]);
                hreg[2] = fmaf(kv, W1s[n][jb + 2], hreg[2]);
                hreg[3] = fmaf(kv, W1s[n][jb + 3], hreg[3]);
            }
        }
    }

    // Write this block's partial (exactly once -> deterministic, no atomics).
    float4 o = make_float4(hreg[0], hreg[1], hreg[2], hreg[3]);
    *(float4*)&partial[(((size_t)split * NTEST) + (m0 + hm)) * 8 + jh * 4] = o;
}

// ---------------------------------------------------------------------------
// Kernel C: reduce 16 split-partials + tiny tanh MLP head.
// ---------------------------------------------------------------------------
__global__ __launch_bounds__(256)
void mlp_head_kernel(const float* __restrict__ partial,   // [16][4096][8]
                     const float* __restrict__ b1,        // [8]
                     const float* __restrict__ W2,        // [8][4]
                     const float* __restrict__ b2,        // [4]
                     const float* __restrict__ W3,        // [4]
                     const float* __restrict__ b3,        // [1]
                     float* __restrict__ out)             // [4096]
{
    const int row = blockIdx.x * 256 + threadIdx.x;
    if (row >= NTEST) return;

    float pre[8] = {0.f, 0.f, 0.f, 0.f, 0.f, 0.f, 0.f, 0.f};
    for (int s = 0; s < 16; ++s) {
        const float* p = partial + (((size_t)s * NTEST) + row) * 8;
        float4 a = *(const float4*)p;
        float4 b = *(const float4*)(p + 4);
        pre[0] += a.x; pre[1] += a.y; pre[2] += a.z; pre[3] += a.w;
        pre[4] += b.x; pre[5] += b.y; pre[6] += b.z; pre[7] += b.w;
    }
    float h1[8];
    #pragma unroll
    for (int j = 0; j < 8; ++j) h1[j] = tanhf(pre[j] + b1[j]);

    float h2[4];
    #pragma unroll
    for (int j2 = 0; j2 < 4; ++j2) {
        float t = b2[j2];
        #pragma unroll
        for (int j = 0; j < 8; ++j) t = fmaf(h1[j], W2[j * 4 + j2], t);
        h2[j2] = tanhf(t);
    }
    float o = b3[0];
    #pragma unroll
    for (int j2 = 0; j2 < 4; ++j2) o = fmaf(h2[j2], W3[j2], o);
    out[row] = o;
}

// ---------------------------------------------------------------------------
// Launch: workspace layout (all re-written every call, deterministic):
//   [0,       4 MiB)  Xb   bf16 4096x512
//   [4 MiB,  12 MiB)  XTb  bf16 8192x512
//   [12 MiB, +16 KiB) nX
//   [+16KiB, +48KiB)  nXT
//   [12 MiB + 64 KiB, +2 MiB) partial f32 [16][4096][8]
// ---------------------------------------------------------------------------
extern "C" void kernel_launch(void* const* d_in, const int* in_sizes, int n_in,
                              void* d_out, int out_size, void* d_ws, size_t ws_size,
                              hipStream_t stream) {
    const float* X  = (const float*)d_in[0];
    const float* XT = (const float*)d_in[1];
    const float* W1 = (const float*)d_in[2];
    const float* b1 = (const float*)d_in[3];
    const float* W2 = (const float*)d_in[4];
    const float* b2 = (const float*)d_in[5];
    const float* W3 = (const float*)d_in[6];
    const float* b3 = (const float*)d_in[7];
    float* out = (float*)d_out;

    char* ws = (char*)d_ws;
    unsigned short* Xb  = (unsigned short*)(ws);
    unsigned short* XTb = (unsigned short*)(ws + (size_t)(4u << 20));
    float* nX      = (float*)(ws + (size_t)(12u << 20));
    float* nXT     = (float*)(ws + (size_t)(12u << 20) + (16u << 10));
    float* partial = (float*)(ws + (size_t)(12u << 20) + (64u << 10));

    cvt_norm_kernel<<<NTEST,  128, 0, stream>>>(X,  Xb,  nX);
    cvt_norm_kernel<<<NTRAIN, 128, 0, stream>>>(XT, XTb, nXT);

    dim3 grid(NTEST / 128, 16);
    rbf_wmma_kernel<<<grid, 256, 0, stream>>>(Xb, XTb, nX, nXT, W1, partial);

    mlp_head_kernel<<<NTEST / 256, 256, 0, stream>>>(partial, b1, W2, b2, W3, b3, out);
}
